// VQLayer_43568148250914
// MI455X (gfx1250) — compile-verified, matrix operand order checked by reading, not run on hardware
//
#include <hip/hip_runtime.h>
#include <hip/hip_bf16.h>

typedef __attribute__((ext_vector_type(16))) __bf16 v16bf;
typedef __attribute__((ext_vector_type(8)))  float  v8f;

// Problem constants (from reference setup_inputs)
// x: [32, 256, 64, 64], conv_w: [256,256], conv_b: [256], codebook: [512,64]
// encoded (pixel-shuffled): [32, 64, 128, 128]
#define NB      32
#define CIN     256
#define HH      64
#define WW      64
#define HWIMG   4096      // 64*64
#define OCH     256
#define DCH     64
#define NIMG    16384     // 128*128
#define NCODES  512
#define EDIM    64

// K-pair offset within a 32-wide K-step for the 16-bit WMMA A/B register layout:
// VGPR i in 0..3 -> K = 2*i + 8*half ; VGPR i in 4..7 -> K = 16 + 2*(i-4) + 8*half
__device__ __forceinline__ int kpair(int i, int half) {
    return ((i < 4) ? (2 * i) : (16 + 2 * (i - 4))) + 8 * half;
}

// ---------------------------------------------------------------------------
// Kernel 1: 1x1 conv (GEMM, M=pixels, N=out-channels, K=256) fused with
// pixel-shuffle scatter into encoded[B,64,128,128].
// One wave computes a 16-pixel x 64-channel tile (4 WMMA N-tiles sharing A).
// ---------------------------------------------------------------------------
__global__ __launch_bounds__(256)
void conv_ps_kernel(const float* __restrict__ x,
                    const float* __restrict__ wgt,
                    const float* __restrict__ bias,
                    float* __restrict__ enc)
{
    __shared__ __bf16 sA[8 * 512];   // per-wave 16x32 bf16 A tile (1KB each)

    const int wave  = threadIdx.x >> 5;
    const int lane  = threadIdx.x & 31;
    const int wid   = blockIdx.x * 8 + wave;        // 0..32767
    const int ptile = wid >> 2;                      // 0..8191
    const int n0    = (wid & 3) * 64;                // out-channel base
    const int pbase = ptile * 16;
    const int b     = pbase >> 12;                   // /4096
    const int hw0   = pbase & 4095;

    __bf16* lA = sA + wave * 512;
    const int m    = lane & 15;
    const int half = lane >> 4;

    const v8f vz = {0.f,0.f,0.f,0.f,0.f,0.f,0.f,0.f};
    v8f acc[4] = {vz, vz, vz, vz};

    for (int k0 = 0; k0 < CIN; k0 += 32) {
        __syncthreads();  // protect WAR on lA vs previous iteration's reads
        // Stage A: lane loads channel c=k0+lane, 16 contiguous pixels (64B run)
        {
            const float* src = x + (((size_t)b * CIN + (k0 + lane)) << 12) + hw0;
            const float4* s4 = (const float4*)src;
            float4 f0 = s4[0], f1 = s4[1], f2 = s4[2], f3 = s4[3];
            float tmp[16];
            *(float4*)(tmp + 0)  = f0;
            *(float4*)(tmp + 4)  = f1;
            *(float4*)(tmp + 8)  = f2;
            *(float4*)(tmp + 12) = f3;
            #pragma unroll
            for (int mm = 0; mm < 16; ++mm)
                lA[mm * 32 + lane] = (__bf16)tmp[mm];
        }
        __syncthreads();

        // Gather A into the 16-bit WMMA A layout (lane = M row, K pairs)
        v16bf a;
        #pragma unroll
        for (int i = 0; i < 8; ++i) {
            int kk = kpair(i, half);
            a[2 * i]     = lA[m * 32 + kk];
            a[2 * i + 1] = lA[m * 32 + kk + 1];
        }

        // 4 N-tiles reuse A
        #pragma unroll
        for (int nt = 0; nt < 4; ++nt) {
            const int ocol = n0 + nt * 16 + m;       // B column = lane&15
            const float* wr = wgt + (size_t)ocol * CIN + k0;
            v16bf bm;
            #pragma unroll
            for (int i = 0; i < 8; ++i) {
                int kk = kpair(i, half);
                float2 w2 = *(const float2*)(wr + kk);
                bm[2 * i]     = (__bf16)w2.x;
                bm[2 * i + 1] = (__bf16)w2.y;
            }
            acc[nt] = __builtin_amdgcn_wmma_f32_16x16x32_bf16(
                false, a, false, bm, (short)0, acc[nt], false, false);
        }
    }

    // Write with fused pixel shuffle: o = d*4 + r0*2 + r1 ; (y,x)=(2h+r0, 2w+r1)
    const int h  = hw0 >> 6;
    const int w0 = hw0 & 63;
    #pragma unroll
    for (int nt = 0; nt < 4; ++nt) {
        const int o  = n0 + nt * 16 + m;             // C column = lane&15
        const int d  = o >> 2;
        const int r0 = (o >> 1) & 1;
        const int r1 = o & 1;
        const float bv = bias[o];
        #pragma unroll
        for (int r = 0; r < 8; ++r) {
            const int mm  = r + 8 * half;            // C row
            const int wc  = w0 + mm;
            const float v = acc[nt][r] + bv;
            size_t oidx = (((size_t)b * DCH + d) * NIMG)
                        + (size_t)(2 * h + r0) * 128 + (2 * wc + r1);
            enc[oidx] = v;
        }
    }
}

// ---------------------------------------------------------------------------
// Kernel 2: VQ — cross = codebook @ encoded (M=codes, K=64, N=pixels),
// argmin_k (|c_k|^2 - 2*cross) per pixel, gather embeddings, write outputs.
// Block stages whole codebook (bf16) + |c|^2 in LDS; one wave per 16 pixels.
// ---------------------------------------------------------------------------
__global__ __launch_bounds__(256)
void vq_kernel(const float* __restrict__ enc,
               const float* __restrict__ cb,
               float* __restrict__ out0,
               float* __restrict__ emb,
               float* __restrict__ idxOut)
{
    __shared__ __bf16 sCB[NCODES * EDIM];  // 64KB
    __shared__ float  sC2[NCODES];         // 2KB
    __shared__ __bf16 sB[8 * 1024];        // 8 waves * (16 px * 64 dims) = 16KB

    const int wave = threadIdx.x >> 5;
    const int lane = threadIdx.x & 31;

    // Stage codebook as bf16 + code norms
    for (int i = threadIdx.x; i < NCODES * EDIM; i += 256)
        sCB[i] = (__bf16)cb[i];
    for (int k = threadIdx.x; k < NCODES; k += 256) {
        float s = 0.f;
        #pragma unroll 8
        for (int d = 0; d < EDIM; ++d) { float v = cb[k * EDIM + d]; s += v * v; }
        sC2[k] = s;
    }
    __syncthreads();

    const int ptile = blockIdx.x * 8 + wave;         // 0..32767
    const int pbase = ptile * 16;
    const int b     = pbase >> 14;                    // /16384
    const int nimg  = pbase & 16383;
    __bf16* lB = sB + wave * 1024;

    // Stage B: 16 pixels x 64 dims, coalesced (pixels contiguous per dim)
    {
        const float* s0 = enc + (((size_t)b * DCH + lane)      << 14) + nimg;
        const float* s1 = enc + (((size_t)b * DCH + lane + 32) << 14) + nimg;
        #pragma unroll
        for (int n = 0; n < 16; ++n) {
            lB[n * 64 + lane]      = (__bf16)s0[n];
            lB[n * 64 + lane + 32] = (__bf16)s1[n];
        }
    }
    __syncthreads();

    const int m    = lane & 15;
    const int half = lane >> 4;

    // Gather B registers once (fixed across all code tiles): 2 K-steps
    v16bf bm[2];
    #pragma unroll
    for (int ks = 0; ks < 2; ++ks) {
        #pragma unroll
        for (int i = 0; i < 8; ++i) {
            int kk = ks * 32 + kpair(i, half);
            bm[ks][2 * i]     = lB[m * 64 + kk];
            bm[ks][2 * i + 1] = lB[m * 64 + kk + 1];
        }
    }

    const v8f vz = {0.f,0.f,0.f,0.f,0.f,0.f,0.f,0.f};
    float minVal = 3.4e38f;
    int   minIdx = 0;

    for (int kt = 0; kt < NCODES / 16; ++kt) {
        v8f acc = vz;
        const __bf16* crow = sCB + (kt * 16 + m) * EDIM;  // A row = lane&15
        #pragma unroll
        for (int ks = 0; ks < 2; ++ks) {
            v16bf a;
            #pragma unroll
            for (int i = 0; i < 8; ++i) {
                int kk = ks * 32 + kpair(i, half);
                a[2 * i]     = crow[kk];
                a[2 * i + 1] = crow[kk + 1];
            }
            acc = __builtin_amdgcn_wmma_f32_16x16x32_bf16(
                false, a, false, bm[ks], (short)0, acc, false, false);
        }
        #pragma unroll
        for (int r = 0; r < 8; ++r) {
            const int code = kt * 16 + r + 8 * half;      // C row
            const float score = sC2[code] - 2.0f * acc[r];
            if (score < minVal) { minVal = score; minIdx = code; }
        }
    }

    // Combine the two lane-halves (rows 0..7 vs 8..15 of each tile)
    {
        float ov = __shfl_xor(minVal, 16, 32);
        int   oi = __shfl_xor(minIdx, 16, 32);
        if (ov < minVal || (ov == minVal && oi < minIdx)) { minVal = ov; minIdx = oi; }
    }

    // Outputs: indices + embeddings (out == embeddings in forward)
    const int p    = pbase + m;
    const int nloc = nimg + m;
    if (half == 0) idxOut[p] = (float)minIdx;

    const float* crowf = cb + (size_t)minIdx * EDIM;
    const size_t obase = ((size_t)b * DCH) * NIMG + (size_t)nloc;
    #pragma unroll
    for (int dd = 0; dd < 32; ++dd) {
        const int d = half * 32 + dd;
        const float v = crowf[d];
        const size_t o = obase + (size_t)d * NIMG;
        out0[o] = v;
        emb[o]  = v;
    }
}

extern "C" void kernel_launch(void* const* d_in, const int* in_sizes, int n_in,
                              void* d_out, int out_size, void* d_ws, size_t ws_size,
                              hipStream_t stream)
{
    const float* x    = (const float*)d_in[0];   // [32,256,64,64]
    const float* wgt  = (const float*)d_in[1];   // [256,256]
    const float* bias = (const float*)d_in[2];   // [256]
    const float* cb   = (const float*)d_in[3];   // [512,64]

    float* outp = (float*)d_out;
    const size_t ENC_N = (size_t)NB * DCH * NIMG;  // 33,554,432
    float* out0 = outp;                 // tuple[0] out
    float* emb  = outp + ENC_N;         // tuple[1] embeddings
    float* enc  = outp + 2 * ENC_N;     // tuple[2] encoded (pixel-shuffled)
    float* idx  = outp + 3 * ENC_N;     // tuple[3] indices (as float)

    // Kernel 1: 32768 waves (8192 pixel-tiles x 4 N-quads), 8 waves/block
    conv_ps_kernel<<<4096, 256, 0, stream>>>(x, wgt, bias, enc);
    // Kernel 2: 32768 pixel-tiles, 8 waves/block
    vq_kernel<<<4096, 256, 0, stream>>>(enc, cb, out0, emb, idx);
}
